// Mamba_3dglobal_36326833389647
// MI455X (gfx1250) — matrix operand order repaired and checked by reading, not compiled, not run on hardware
//
#include <hip/hip_runtime.h>
#include <math.h>

// ---- problem constants -----------------------------------------------------
#define DIMC   1024
#define DMODEL 256
#define DINNER 512
#define DSTATE 16
#define DTRANK 16
#define DCONV  4
#define HW     24
#define HW2    48
#define LSEQ   2305                 // 1 global token + 48*48 spatial tokens
#define BATCH  4
#define MTOT   (BATCH * LSEQ)       // 9220 valid rows
#define MPAD   9248                 // 32 * 289 (M padded to 32-row GEMM tile)
#define MT32   (MPAD / 32)          // 289 row tiles

// ---- WMMA types ------------------------------------------------------------
typedef __attribute__((ext_vector_type(16))) __bf16 bf16x16;
typedef __attribute__((ext_vector_type(8)))  float  f32x8;

union Frag {          // 16 bf16 values = two 16-byte chunks
  bf16x16 v;
  uint4   q[2];
};

// A-operand: per lane, K-chunks at +0 and +16 elements (ISA 16-bit A layout)
__device__ __forceinline__ void loadA(Frag& f, const unsigned short* p) {
  f.q[0] = *(const uint4*)(p);
  f.q[1] = *(const uint4*)(p + 16);
}
// B-operand: per lane, K-chunks at +0 and +16 elements (same as A layout)
__device__ __forceinline__ void loadB(Frag& f, const unsigned short* p) {
  f.q[0] = *(const uint4*)(p);
  f.q[1] = *(const uint4*)(p + 16);
}

__device__ __forceinline__ unsigned short f2bf(float f) {  // RNE f32->bf16
  unsigned int u = __float_as_uint(f);
  u += 0x7fffu + ((u >> 16) & 1u);
  return (unsigned short)(u >> 16);
}

__device__ __forceinline__ float silu(float x) { return x / (1.f + expf(-x)); }

// ---- 1) build seq (pixel_shuffle + token concat) as bf16 [MPAD, 256] -------
__global__ void pack_seq_kernel(const float* __restrict__ x,
                                const float* __restrict__ gt,
                                unsigned short* __restrict__ seq) {
  int t = blockIdx.x * blockDim.x + threadIdx.x;
  if (t >= MPAD * DMODEL) return;
  int m = t / DMODEL;
  int c = t - m * DMODEL;
  float v = 0.f;
  if (m < MTOT) {
    int b = m / LSEQ;
    int l = m - b * LSEQ;
    if (l == 0) {
      v = gt[c];
    } else {
      int p  = l - 1;
      int h2 = p / HW2, w2 = p - h2 * HW2;
      int ch = c * 4 + (h2 & 1) * 2 + (w2 & 1);
      v = x[((b * DIMC + ch) * HW + (h2 >> 1)) * HW + (w2 >> 1)];
    }
  }
  seq[t] = f2bf(v);
}

// ---- weight f32 -> bf16 (with optional zero row padding) -------------------
__global__ void pack_w_kernel(const float* __restrict__ src,
                              unsigned short* __restrict__ dst,
                              int rows_valid, int rows_total, int cols) {
  int t = blockIdx.x * blockDim.x + threadIdx.x;
  if (t >= rows_total * cols) return;
  int r = t / cols;
  int c = t - r * cols;
  float v = (r < rows_valid) ? src[r * cols + c] : 0.f;
  dst[t] = f2bf(v);
}

// ---- bf16 WMMA GEMM: C[M,N] = A[M,K] * W[N,K]^T ----------------------------
// one wave per block; wave computes a 32(M) x 64(N) tile, K in steps of 32,
// register double-buffered so loads for step k+32 overlap WMMAs of step k.
__global__ void gemm_wmma_kernel(const unsigned short* __restrict__ A,
                                 const unsigned short* __restrict__ W,
                                 float* __restrict__ C,
                                 int K, int ldc) {
  const int m0   = blockIdx.x * 32;
  const int n0   = blockIdx.y * 64;
  const int lane = threadIdx.x & 31;
  const int half = lane >> 4;
  const int lrow = lane & 15;

  const unsigned short* arow0 = A + (size_t)(m0 + lrow) * K + half * 8;
  const unsigned short* arow1 = arow0 + (size_t)16 * K;
  const unsigned short* wbase = W + (size_t)(n0 + lrow) * K + half * 8;

  f32x8 acc[2][4] = {};
  Frag a0[2], b0[4];
  loadA(a0[0], arow0);
  loadA(a0[1], arow1);
#pragma unroll
  for (int j = 0; j < 4; ++j) loadB(b0[j], wbase + (size_t)j * 16 * K);

  for (int k0 = 0; k0 < K; k0 += 32) {
    // prefetch next K step (wraps to 0 on final iteration; result unused)
    const int kn = (k0 + 32 < K) ? k0 + 32 : 0;
    Frag a1[2], b1[4];
    loadA(a1[0], arow0 + kn);
    loadA(a1[1], arow1 + kn);
#pragma unroll
    for (int j = 0; j < 4; ++j) loadB(b1[j], wbase + (size_t)j * 16 * K + kn);

#pragma unroll
    for (int mi = 0; mi < 2; ++mi)
#pragma unroll
      for (int j = 0; j < 4; ++j)
        acc[mi][j] = __builtin_amdgcn_wmma_f32_16x16x32_bf16(
            false, a0[mi].v, false, b0[j].v, (short)0, acc[mi][j], false, false);

    a0[0] = a1[0]; a0[1] = a1[1];
#pragma unroll
    for (int j = 0; j < 4; ++j) b0[j] = b1[j];
  }

#pragma unroll
  for (int mi = 0; mi < 2; ++mi)
#pragma unroll
    for (int j = 0; j < 4; ++j) {
      int n = n0 + j * 16 + lrow;
#pragma unroll
      for (int i = 0; i < 8; ++i) {
        int mr = m0 + mi * 16 + half * 8 + i;
        C[(size_t)mr * ldc + n] = acc[mi][j][i];
      }
    }
}

// ---- same GEMM but store scatters directly into pixel_unshuffled output ----
__global__ void gemm_wmma_scatter_kernel(const unsigned short* __restrict__ A,
                                         const unsigned short* __restrict__ W,
                                         float* __restrict__ out,
                                         int K) {
  const int m0   = blockIdx.x * 32;
  const int n0   = blockIdx.y * 64;
  const int lane = threadIdx.x & 31;
  const int half = lane >> 4;
  const int lrow = lane & 15;

  const unsigned short* arow0 = A + (size_t)(m0 + lrow) * K + half * 8;
  const unsigned short* arow1 = arow0 + (size_t)16 * K;
  const unsigned short* wbase = W + (size_t)(n0 + lrow) * K + half * 8;

  f32x8 acc[2][4] = {};
  Frag a0[2], b0[4];
  loadA(a0[0], arow0);
  loadA(a0[1], arow1);
#pragma unroll
  for (int j = 0; j < 4; ++j) loadB(b0[j], wbase + (size_t)j * 16 * K);

  for (int k0 = 0; k0 < K; k0 += 32) {
    const int kn = (k0 + 32 < K) ? k0 + 32 : 0;
    Frag a1[2], b1[4];
    loadA(a1[0], arow0 + kn);
    loadA(a1[1], arow1 + kn);
#pragma unroll
    for (int j = 0; j < 4; ++j) loadB(b1[j], wbase + (size_t)j * 16 * K + kn);

#pragma unroll
    for (int mi = 0; mi < 2; ++mi)
#pragma unroll
      for (int j = 0; j < 4; ++j)
        acc[mi][j] = __builtin_amdgcn_wmma_f32_16x16x32_bf16(
            false, a0[mi].v, false, b0[j].v, (short)0, acc[mi][j], false, false);

    a0[0] = a1[0]; a0[1] = a1[1];
#pragma unroll
    for (int j = 0; j < 4; ++j) b0[j] = b1[j];
  }

#pragma unroll
  for (int mi = 0; mi < 2; ++mi)
#pragma unroll
    for (int j = 0; j < 4; ++j) {
      int n = n0 + j * 16 + lrow;        // model channel c in [0,256)
#pragma unroll
      for (int i = 0; i < 8; ++i) {
        int mr = m0 + mi * 16 + half * 8 + i;
        if (mr < MTOT) {
          int b = mr / LSEQ;
          int l = mr - b * LSEQ;
          if (l > 0) {                   // drop global token
            int p  = l - 1;
            int h2 = p / HW2, w2 = p - h2 * HW2;
            int ch = n * 4 + (h2 & 1) * 2 + (w2 & 1);
            out[((b * DIMC + ch) * HW + (h2 >> 1)) * HW + (w2 >> 1)] = acc[mi][j][i];
          }
        }
      }
    }
}

// ---- causal depthwise conv(4) + bias + SiLU --------------------------------
__global__ void conv_silu_kernel(const float* __restrict__ xz,      // [MPAD,1024]
                                 const float* __restrict__ conv_w,  // [512,4]
                                 const float* __restrict__ conv_b,  // [512]
                                 float* __restrict__ u32,           // [MPAD,512]
                                 unsigned short* __restrict__ u16) {
  int t = blockIdx.x * blockDim.x + threadIdx.x;
  if (t >= MPAD * DINNER) return;
  int m = t / DINNER;
  int d = t - m * DINNER;
  int b = m / LSEQ;
  int l = m - b * LSEQ;
  float acc = conv_b[d];
#pragma unroll
  for (int k = 0; k < DCONV; ++k) {
    int ll = l - (DCONV - 1) + k;
    if (ll >= 0 && m - (DCONV - 1) + k < MPAD)
      acc += xz[(size_t)(m - (DCONV - 1) + k) * DIMC + d] * conv_w[d * DCONV + k];
  }
  float s = silu(acc);
  u32[t] = s;
  u16[t] = f2bf(s);
}

// ---- fused selective scan: dt_proj+softplus, recurrence, D-skip, z-gate ----
// one lane per (b,d); 16-element state in VGPRs; per-step dt/B/C row via LDS.
__global__ void scan_kernel(const float* __restrict__ x_dbl,  // [MPAD,64] dt|B|C|pad
                            const float* __restrict__ u32,    // [MPAD,512]
                            const float* __restrict__ xz,     // [MPAD,1024] (z at +512)
                            const float* __restrict__ dt_w,   // [512,16]
                            const float* __restrict__ dt_b,   // [512]
                            const float* __restrict__ A_log,  // [512,16]
                            const float* __restrict__ Dvec,   // [512]
                            unsigned short* __restrict__ y16) {
  __shared__ float srow[64];
  const int b = blockIdx.x >> 1;
  const int d = ((blockIdx.x & 1) << 8) + threadIdx.x;

  float h[DSTATE], Ar[DSTATE], dtw[DSTATE];
#pragma unroll
  for (int n = 0; n < DSTATE; ++n) {
    h[n]   = 0.f;
    Ar[n]  = -expf(A_log[d * DSTATE + n]);
    dtw[n] = dt_w[d * DSTATE + n];
  }
  const float bias = dt_b[d];
  const float Dd   = Dvec[d];

  for (int l = 0; l < LSEQ; ++l) {
    const int m = b * LSEQ + l;
    if (threadIdx.x < 64) srow[threadIdx.x] = x_dbl[(size_t)m * 64 + threadIdx.x];
    __syncthreads();

    float dtr = bias;
#pragma unroll
    for (int r = 0; r < DTRANK; ++r) dtr += srow[r] * dtw[r];
    float dt = (dtr > 20.f) ? dtr : log1pf(expf(dtr));   // softplus

    float ut = u32[(size_t)m * DINNER + d];
    float du = dt * ut;
    float y  = 0.f;
#pragma unroll
    for (int n = 0; n < DSTATE; ++n) {
      float dA = expf(dt * Ar[n]);
      h[n] = dA * h[n] + du * srow[DTRANK + n];
      y   += h[n] * srow[DTRANK + DSTATE + n];
    }
    float z  = xz[(size_t)m * DIMC + DINNER + d];
    float yf = (y + ut * Dd) * silu(z);
    y16[(size_t)m * DINNER + d] = f2bf(yf);
    __syncthreads();
  }
}

// ---- host side -------------------------------------------------------------
extern "C" void kernel_launch(void* const* d_in, const int* in_sizes, int n_in,
                              void* d_out, int out_size, void* d_ws, size_t ws_size,
                              hipStream_t stream) {
  const float* x      = (const float*)d_in[0];
  const float* gt     = (const float*)d_in[1];
  const float* in_w   = (const float*)d_in[2];   // (1024,256)
  const float* conv_w = (const float*)d_in[3];   // (512,4)
  const float* conv_b = (const float*)d_in[4];   // (512)
  const float* xp_w   = (const float*)d_in[5];   // (48,512)
  const float* dt_w   = (const float*)d_in[6];   // (512,16)
  const float* dt_b   = (const float*)d_in[7];   // (512)
  const float* A_log  = (const float*)d_in[8];   // (512,16)
  const float* Dv     = (const float*)d_in[9];   // (512)
  const float* out_w  = (const float*)d_in[10];  // (256,512)
  float* out = (float*)d_out;

  char* ws = (char*)d_ws;
  size_t off = 0;
  auto alloc = [&](size_t bytes) -> void* {
    void* p = ws + off;
    off = (off + bytes + 255) & ~(size_t)255;
    return p;
  };
  unsigned short* seq16 = (unsigned short*)alloc((size_t)MPAD * DMODEL * 2);
  unsigned short* w_in  = (unsigned short*)alloc((size_t)DIMC * DMODEL * 2);
  unsigned short* w_x   = (unsigned short*)alloc((size_t)64 * DINNER * 2);
  unsigned short* w_out = (unsigned short*)alloc((size_t)DMODEL * DINNER * 2);
  float*          xz    = (float*)alloc((size_t)MPAD * DIMC * 4);
  float*          u32   = (float*)alloc((size_t)MPAD * DINNER * 4);
  unsigned short* u16   = (unsigned short*)alloc((size_t)MPAD * DINNER * 2);
  float*          xdbl  = (float*)alloc((size_t)MPAD * 64 * 4);
  unsigned short* y16   = (unsigned short*)alloc((size_t)MPAD * DINNER * 2);

  pack_seq_kernel<<<(MPAD * DMODEL + 255) / 256, 256, 0, stream>>>(x, gt, seq16);
  pack_w_kernel<<<(DIMC * DMODEL + 255) / 256, 256, 0, stream>>>(in_w, w_in, DIMC, DIMC, DMODEL);
  pack_w_kernel<<<(64 * DINNER + 255) / 256, 256, 0, stream>>>(xp_w, w_x, 48, 64, DINNER);
  pack_w_kernel<<<(DMODEL * DINNER + 255) / 256, 256, 0, stream>>>(out_w, w_out, DMODEL, DMODEL, DINNER);

  // in_proj: [MPAD,256] x [1024,256]^T -> xz [MPAD,1024]
  gemm_wmma_kernel<<<dim3(MT32, DIMC / 64), 32, 0, stream>>>(seq16, w_in, xz, DMODEL, DIMC);
  conv_silu_kernel<<<(MPAD * DINNER + 255) / 256, 256, 0, stream>>>(xz, conv_w, conv_b, u32, u16);
  // x_proj: [MPAD,512] x [64,512]^T -> x_dbl [MPAD,64]
  gemm_wmma_kernel<<<dim3(MT32, 1), 32, 0, stream>>>(u16, w_x, xdbl, DINNER, 64);
  // sequential scan over L, parallel over 4*512 channels
  scan_kernel<<<8, 256, 0, stream>>>(xdbl, u32, xz, dt_w, dt_b, A_log, Dv, y16);
  // out_proj + drop token + pixel_unshuffle scatter into d_out
  gemm_wmma_scatter_kernel<<<dim3(MT32, DMODEL / 64), 32, 0, stream>>>(y16, w_out, out, DINNER);
}
